// CoNHD_GD_Layer_87282325389907
// MI455X (gfx1250) — matrix operand order repaired and checked by reading, not compiled, not run on hardware
//
#include <hip/hip_runtime.h>

typedef __attribute__((ext_vector_type(16))) _Float16 v16h;
typedef __attribute__((ext_vector_type(8)))  _Float16 v8h;
typedef __attribute__((ext_vector_type(4)))  _Float16 v4h;
typedef __attribute__((ext_vector_type(8)))  float    v8f;
typedef __attribute__((ext_vector_type(4)))  float    v4f;

// problem sizes
constexpr int EN  = 65536;  // E edges
constexpr int DD  = 256;    // feature dim D
constexpr int WDm = 64;     // weight dim
// v side: 2048 nodes x 32 ; e side: 4096 hyperedges x 16

// epilogue flag bits (regular GEMM)
enum {
  F_BIAS   = 1,
  F_RELU   = 2,
  F_RESID  = 4,
  F_OUTF16 = 8
};

// fragment index helpers for the 16-bit WMMA A-operand layout:
// lane (g = lane>>4) holds halves h=0..15 with K = (h&7) + (h>=8?16:0) + g*8.
// inverse: K-col c in [0,32) -> g = (c>>3)&1 ; h = (c&7) | ((c>>1)&8)
__device__ __forceinline__ int frag_g(int c) { return (c >> 3) & 1; }
__device__ __forceinline__ int frag_h(int c) { return (c & 7) | ((c >> 1) & 8); }

// ---------------------------------------------------------------------------
// Pack an f32 [K, 256] weight into WMMA B-operand layout (f16):
// block (kt,nt): 32 lanes x 16 halves; lane: N = nt*16+(lane&15),
// halves h: K = kt*32 + (lane>>4)*8 + (h&7) + (h>=8?16:0)   (A-fragment K order).
// ---------------------------------------------------------------------------
__global__ void pack_w_kernel(const float* __restrict__ W, _Float16* __restrict__ Wp,
                              int K, int Nc) {
  int idx = blockIdx.x * 256 + threadIdx.x;
  if (idx >= K * Nc) return;
  int h    = idx & 15;
  int lane = (idx >> 4) & 31;
  int blk  = idx >> 9;
  int numNt = Nc >> 4;
  int kt = blk / numNt;
  int nt = blk - kt * numNt;
  int g  = lane >> 4;
  int k  = kt * 32 + g * 8 + (h & 7) + ((h >> 3) << 4);
  int n  = nt * 16 + (lane & 15);
  Wp[idx] = (_Float16)W[(size_t)k * Nc + n];
}

__global__ void inv_perm_kernel(const int* __restrict__ perm, int* __restrict__ inv, int n) {
  int i = blockIdx.x * 256 + threadIdx.x;
  if (i < n) inv[perm[i]] = i;
}

__global__ void zero_h_kernel(_Float16* p, int count) {
  int i = blockIdx.x * 256 + threadIdx.x;
  if (i < count) p[i] = (_Float16)0.0f;
}

// ---------------------------------------------------------------------------
// Tiled WMMA GEMM: C[EN, 256] = epi( A[EN, K](f32) @ Wp(K x 256, packed f16) )
// Block tile 128x128, 8 waves (4 x 2), wave tile 32x64 = 2x4 accumulators.
// Double-buffered LDS A tile in fragment-native layout; staging is
// float4 global loads -> f16 convert -> b64 LDS stores.
// ---------------------------------------------------------------------------
template<int FLAGS, int KDIM>
__global__ __launch_bounds__(256, 1)
void gemm_wmma_kernel(const float* __restrict__ A, int lda,
                      const _Float16* __restrict__ Wp,
                      const float* __restrict__ bias,
                      const float* __restrict__ R,
                      float* __restrict__ Cout, _Float16* __restrict__ C16) {
  constexpr int NKT = KDIM / 32;
  __shared__ _Float16 sA[2][128 * 32];  // [buf][mtile(8)][lane(32)][h(16)]

  const int t    = threadIdx.x;
  const int lane = t & 31;
  const int w    = t >> 5;
  const int wm   = w & 3;
  const int wn   = w >> 2;
  const int mbase = blockIdx.x * 128;
  const int nbase = blockIdx.y * 128;

  // staging geometry: 4 float4 quads per thread (128 rows x 32 cols)
  int qrow[4], qcol[4];
#pragma unroll
  for (int i = 0; i < 4; ++i) { int qi = i * 256 + t; qrow[i] = qi >> 3; qcol[i] = (qi & 7) * 4; }

  v4f regs[4];
  auto loadRegs = [&](int kt) {
#pragma unroll
    for (int i = 0; i < 4; ++i)
      regs[i] = *(const v4f*)&A[(size_t)(mbase + qrow[i]) * lda + kt * 32 + qcol[i]];
  };
  auto storeLDS = [&](_Float16* buf) {
#pragma unroll
    for (int i = 0; i < 4; ++i) {
      int c0 = qcol[i];
      int g  = frag_g(c0);
      int h0 = frag_h(c0);  // contiguous 4 halves for an aligned quad
      v4h hv;
#pragma unroll
      for (int j = 0; j < 4; ++j) hv[j] = (_Float16)regs[i][j];
      *(v4h*)&buf[(((qrow[i] >> 4) * 32) + g * 16 + (qrow[i] & 15)) * 16 + h0] = hv;
    }
  };

  const v8f z8 = {0.f, 0.f, 0.f, 0.f, 0.f, 0.f, 0.f, 0.f};
  v8f acc[2][4];
#pragma unroll
  for (int i = 0; i < 2; ++i)
#pragma unroll
    for (int j = 0; j < 4; ++j) acc[i][j] = z8;

  loadRegs(0);
#pragma unroll 2
  for (int kt = 0; kt < NKT; ++kt) {
    _Float16* buf = sA[kt & 1];
    storeLDS(buf);
    __syncthreads();
    if (kt + 1 < NKT) loadRegs(kt + 1);  // prefetch next chunk during WMMAs

    v16h afrag[2];
#pragma unroll
    for (int mi = 0; mi < 2; ++mi)
      afrag[mi] = *(const v16h*)&buf[((wm * 2 + mi) * 32 + lane) * 16];

#pragma unroll
    for (int ni = 0; ni < 4; ++ni) {
      int ntg = (nbase >> 4) + wn * 4 + ni;
      const v16h b = *(const v16h*)&Wp[((size_t)(kt * (DD / 16) + ntg) * 32 + lane) * 16];
#pragma unroll
      for (int mi = 0; mi < 2; ++mi)
        acc[mi][ni] = __builtin_amdgcn_wmma_f32_16x16x32_f16(
            false, afrag[mi], false, b, (short)0, acc[mi][ni], false, false);
    }
  }

  // epilogue: C layout lane -> (row = r + (lane>>4)*8, col = lane&15)
  const int g  = lane >> 4;
  const int ln = lane & 15;
#pragma unroll
  for (int mi = 0; mi < 2; ++mi) {
#pragma unroll
    for (int ni = 0; ni < 4; ++ni) {
      int col = nbase + wn * 64 + ni * 16 + ln;
#pragma unroll
      for (int r = 0; r < 8; ++r) {
        int row = mbase + wm * 32 + mi * 16 + g * 8 + r;
        float v = acc[mi][ni][r];
        if constexpr (FLAGS & F_BIAS)  v += bias[col];
        if constexpr (FLAGS & F_RELU)  v = v > 0.f ? v : 0.f;
        if constexpr (FLAGS & F_RESID) v += R[(size_t)row * DD + col];
        size_t oi = (size_t)row * DD + col;
        if constexpr (FLAGS & F_OUTF16) C16[oi] = (_Float16)v;
        else                            Cout[oi] = v;
      }
    }
  }
}

// ---------------------------------------------------------------------------
// Fused final update GEMM: out[EN,256] = cat(A0,A1,A2,A3)[E,1024] @ Wp + bias,
// where each 256-wide slice has its own source pointer and optional row-gather
// (perm / inv_perm / identity). K = 1024 -> 32 chunks; source switch per chunk
// is wave-uniform (kt>>3). Avoids 3 read+write accumulate passes over d_out.
// ---------------------------------------------------------------------------
__global__ __launch_bounds__(256, 1)
void gemm_upd_kernel(const float* __restrict__ A0, const int* __restrict__ I0,
                     const float* __restrict__ A1, const int* __restrict__ I1,
                     const float* __restrict__ A2, const int* __restrict__ I2,
                     const float* __restrict__ A3, const int* __restrict__ I3,
                     const _Float16* __restrict__ Wp,
                     const float* __restrict__ bias,
                     float* __restrict__ Cout) {
  constexpr int NKT = 32;  // K = 1024
  __shared__ _Float16 sA[2][128 * 32];

  const int t    = threadIdx.x;
  const int lane = t & 31;
  const int w    = t >> 5;
  const int wm   = w & 3;
  const int wn   = w >> 2;
  const int mbase = blockIdx.x * 128;
  const int nbase = blockIdx.y * 128;

  int qrow[4], qcol[4];
#pragma unroll
  for (int i = 0; i < 4; ++i) { int qi = i * 256 + t; qrow[i] = qi >> 3; qcol[i] = (qi & 7) * 4; }

  v4f regs[4];
  auto loadRegs = [&](int kt) {
    int sel = kt >> 3;  // wave-uniform source select
    const float* As; const int* ix;
    if      (sel == 0) { As = A0; ix = I0; }
    else if (sel == 1) { As = A1; ix = I1; }
    else if (sel == 2) { As = A2; ix = I2; }
    else               { As = A3; ix = I3; }
    int ktl = kt & 7;
#pragma unroll
    for (int i = 0; i < 4; ++i) {
      int grow = mbase + qrow[i];
      if (ix) grow = ix[grow];
      regs[i] = *(const v4f*)&As[(size_t)grow * DD + ktl * 32 + qcol[i]];
    }
  };
  auto storeLDS = [&](_Float16* buf) {
#pragma unroll
    for (int i = 0; i < 4; ++i) {
      int c0 = qcol[i];
      v4h hv;
#pragma unroll
      for (int j = 0; j < 4; ++j) hv[j] = (_Float16)regs[i][j];
      *(v4h*)&buf[(((qrow[i] >> 4) * 32) + frag_g(c0) * 16 + (qrow[i] & 15)) * 16 + frag_h(c0)] = hv;
    }
  };

  const v8f z8 = {0.f, 0.f, 0.f, 0.f, 0.f, 0.f, 0.f, 0.f};
  v8f acc[2][4];
#pragma unroll
  for (int i = 0; i < 2; ++i)
#pragma unroll
    for (int j = 0; j < 4; ++j) acc[i][j] = z8;

  loadRegs(0);
#pragma unroll 2
  for (int kt = 0; kt < NKT; ++kt) {
    _Float16* buf = sA[kt & 1];
    storeLDS(buf);
    __syncthreads();
    if (kt + 1 < NKT) loadRegs(kt + 1);

    v16h afrag[2];
#pragma unroll
    for (int mi = 0; mi < 2; ++mi)
      afrag[mi] = *(const v16h*)&buf[((wm * 2 + mi) * 32 + lane) * 16];

#pragma unroll
    for (int ni = 0; ni < 4; ++ni) {
      int ntg = (nbase >> 4) + wn * 4 + ni;
      const v16h b = *(const v16h*)&Wp[((size_t)(kt * (DD / 16) + ntg) * 32 + lane) * 16];
#pragma unroll
      for (int mi = 0; mi < 2; ++mi)
        acc[mi][ni] = __builtin_amdgcn_wmma_f32_16x16x32_f16(
            false, afrag[mi], false, b, (short)0, acc[mi][ni], false, false);
    }
  }

  const int g  = lane >> 4;
  const int ln = lane & 15;
#pragma unroll
  for (int mi = 0; mi < 2; ++mi) {
#pragma unroll
    for (int ni = 0; ni < 4; ++ni) {
      int col = nbase + wn * 64 + ni * 16 + ln;
#pragma unroll
      for (int r = 0; r < 8; ++r) {
        int row = mbase + wm * 32 + mi * 16 + g * 8 + r;
        Cout[(size_t)row * DD + col] = acc[mi][ni][r] + bias[col];
      }
    }
  }
}

// ---------------------------------------------------------------------------
// Per-set attention: one block per node/hyperedge, one wave per head (H=4).
// NSEQ = 32 (v side) or 16 (e side) compile-time; head dim 64; scale 1/16.
// O = Qh + softmax(Qh Kh^T * scale) Vh, written to O[E,256] f32.
// ---------------------------------------------------------------------------
template<int NSEQ>
__global__ __launch_bounds__(128, 1)
void attn_kernel(const _Float16* __restrict__ Q, const _Float16* __restrict__ Kb,
                 const _Float16* __restrict__ Vb, float* __restrict__ O) {
  constexpr int   MT    = NSEQ / 16;  // 2 (v) or 1 (e)
  constexpr float scale = 0.0625f;
  __shared__ _Float16 sP[4 * 2 * 32 * 16];  // per head: [mtile][lane][h]

  const int t    = threadIdx.x;
  const int lane = t & 31;
  const int head = t >> 5;
  const int g    = lane >> 4;
  const int ln   = lane & 15;
  const int rb   = blockIdx.x * NSEQ;
  const int hc   = head * 64;
  _Float16* sPw  = &sP[head * 2 * 32 * 16];

  const v8f z8 = {0.f, 0.f, 0.f, 0.f, 0.f, 0.f, 0.f, 0.f};
  v8f s[MT][MT];
#pragma unroll
  for (int i = 0; i < MT; ++i)
#pragma unroll
    for (int j = 0; j < MT; ++j) s[i][j] = z8;

  // scores = Qh @ Kh^T  (K-dim = head dim 64 -> 2 WMMA K-steps)
#pragma unroll
  for (int kk = 0; kk < 2; ++kk) {
    v16h a[MT];
#pragma unroll
    for (int mi = 0; mi < MT; ++mi) {
      size_t row = (size_t)(rb + mi * 16 + ln);
      v8h lo = *(const v8h*)&Q[row * DD + hc + kk * 32 + g * 16];
      v8h hi = *(const v8h*)&Q[row * DD + hc + kk * 32 + g * 16 + 8];
#pragma unroll
      for (int x = 0; x < 8; ++x) { a[mi][x] = lo[x]; a[mi][x + 8] = hi[x]; }
    }
#pragma unroll
    for (int nj = 0; nj < MT; ++nj) {
      size_t kr = (size_t)(rb + nj * 16 + ln);
      const v16h b = *(const v16h*)&Kb[kr * DD + hc + kk * 32 + g * 16];
#pragma unroll
      for (int mi = 0; mi < MT; ++mi)
        s[mi][nj] = __builtin_amdgcn_wmma_f32_16x16x32_f16(
            false, a[mi], false, b, (short)0, s[mi][nj], false, false);
    }
  }

  // row-wise softmax (a row lives in the 16 lanes of one half-wave)
  float p[MT][MT][8];
#pragma unroll
  for (int mi = 0; mi < MT; ++mi) {
#pragma unroll
    for (int r = 0; r < 8; ++r) {
      float mx = -1e30f;
#pragma unroll
      for (int nj = 0; nj < MT; ++nj) mx = fmaxf(mx, s[mi][nj][r] * scale);
#pragma unroll
      for (int msk = 1; msk < 16; msk <<= 1) mx = fmaxf(mx, __shfl_xor(mx, msk, 32));
      float sum = 0.f;
#pragma unroll
      for (int nj = 0; nj < MT; ++nj) {
        float ev = __expf(s[mi][nj][r] * scale - mx);
        p[mi][nj][r] = ev;
        sum += ev;
      }
#pragma unroll
      for (int msk = 1; msk < 16; msk <<= 1) sum += __shfl_xor(sum, msk, 32);
      float inv = 1.f / sum;
#pragma unroll
      for (int nj = 0; nj < MT; ++nj) p[mi][nj][r] *= inv;
    }
  }

  // stage probabilities to LDS in A-fragment layout (zero-padded to 32 keys)
#pragma unroll
  for (int i = 0; i < MT * 512 / 32; ++i) sPw[i * 32 + lane] = (_Float16)0.0f;
  __syncthreads();
#pragma unroll
  for (int mi = 0; mi < MT; ++mi)
#pragma unroll
    for (int nj = 0; nj < MT; ++nj)
#pragma unroll
      for (int r = 0; r < 8; ++r) {
        int row = g * 8 + r;          // row within mtile
        int kc  = nj * 16 + ln;       // key column 0..31
        sPw[(mi * 32 + frag_g(kc) * 16 + row) * 16 + frag_h(kc)] = (_Float16)p[mi][nj][r];
      }
  __syncthreads();

  v16h pa[MT];
#pragma unroll
  for (int mi = 0; mi < MT; ++mi)
    pa[mi] = *(const v16h*)&sPw[(mi * 32 + lane) * 16];

  // O = Qh + P @ Vh  (K = 32 keys -> one WMMA step; head dim 64 -> 4 N tiles)
#pragma unroll
  for (int nv = 0; nv < 4; ++nv) {
    int c = hc + nv * 16 + ln;
    v16h bv;
#pragma unroll
    for (int h2 = 0; h2 < 16; ++h2)
      bv[h2] = Vb[(size_t)(rb + g * 8 + (h2 & 7) + ((h2 >> 3) << 4)) * DD + c];
    v8f o[MT];
#pragma unroll
    for (int mi = 0; mi < MT; ++mi) o[mi] = z8;
#pragma unroll
    for (int mi = 0; mi < MT; ++mi)
      o[mi] = __builtin_amdgcn_wmma_f32_16x16x32_f16(
          false, pa[mi], false, bv, (short)0, o[mi], false, false);
#pragma unroll
    for (int mi = 0; mi < MT; ++mi)
#pragma unroll
      for (int r = 0; r < 8; ++r) {
        size_t row = (size_t)(rb + mi * 16 + g * 8 + r);
        size_t oi  = row * DD + c;
        O[oi] = o[mi][r] + (float)Q[oi];
      }
  }
}

// ---------------------------------------------------------------------------
// host side
// ---------------------------------------------------------------------------
template<int FLAGS, int KDIM>
static inline void launch_gemm(hipStream_t stream, const float* A, int lda,
                               const _Float16* Wp,
                               const float* bias, const float* R,
                               float* Cout, _Float16* C16) {
  gemm_wmma_kernel<FLAGS, KDIM><<<dim3(EN / 128, DD / 128), dim3(256), 0, stream>>>(
      A, lda, Wp, bias, R, Cout, C16);
}

extern "C" void kernel_launch(void* const* d_in, const int* in_sizes, int n_in,
                              void* d_out, int out_size, void* d_ws, size_t ws_size,
                              hipStream_t stream) {
  (void)in_sizes; (void)n_in; (void)out_size; (void)ws_size;

  const float* co_in  = (const float*)d_in[0];
  const float* co_con = (const float*)d_in[1];
  const float* co_0   = (const float*)d_in[2];
  const float* w_in   = (const float*)d_in[3];
  const float* w_con  = (const float*)d_in[4];
  const float* peW[2] = {(const float*)d_in[5], (const float*)d_in[7]};
  const float* peB[2] = {(const float*)d_in[6], (const float*)d_in[8]};
  const float* Wq[2]  = {(const float*)d_in[9],  (const float*)d_in[17]};
  const float* Bq[2]  = {(const float*)d_in[10], (const float*)d_in[18]};
  const float* Wk[2]  = {(const float*)d_in[11], (const float*)d_in[19]};
  const float* Bk[2]  = {(const float*)d_in[12], (const float*)d_in[20]};
  const float* Wv[2]  = {(const float*)d_in[13], (const float*)d_in[21]};
  const float* Bv[2]  = {(const float*)d_in[14], (const float*)d_in[22]};
  const float* Wo[2]  = {(const float*)d_in[15], (const float*)d_in[23]};
  const float* Bo[2]  = {(const float*)d_in[16], (const float*)d_in[24]};
  const float* updW   = (const float*)d_in[25];
  const float* updb   = (const float*)d_in[26];
  const int*   perm   = (const int*)d_in[27];

  // carve workspace
  char* ws = (char*)d_ws;
  size_t off = 0;
  auto carve = [&](size_t bytes) -> void* {
    void* p = ws + off;
    off += (bytes + 255) & ~(size_t)255;
    return p;
  };
  float* X  = (float*)carve((size_t)EN * DD * 4);   // current activations
  float* Ob = (float*)carve((size_t)EN * DD * 4);   // attention output O
  float* CV = (float*)carve((size_t)EN * DD * 4);   // co_feat_v
  float* CE = (float*)carve((size_t)EN * DD * 4);   // co_feat_e
  _Float16* Q16 = (_Float16*)carve((size_t)(EN + 16) * DD * 2);
  _Float16* K16 = (_Float16*)carve((size_t)(EN + 16) * DD * 2);
  _Float16* V16 = (_Float16*)carve((size_t)(EN + 16) * DD * 2);
  int* invp = (int*)carve((size_t)EN * 4);
  _Float16* pPE[2];
  pPE[0] = (_Float16*)carve((size_t)WDm * DD * 2);
  pPE[1] = (_Float16*)carve((size_t)WDm * DD * 2);
  _Float16 *pQ[2][2], *pK[2][2], *pV[2][2], *pO[2][2];
  for (int s2 = 0; s2 < 2; ++s2)
    for (int l = 0; l < 2; ++l) {
      pQ[s2][l] = (_Float16*)carve((size_t)DD * DD * 2);
      pK[s2][l] = (_Float16*)carve((size_t)DD * DD * 2);
      pV[s2][l] = (_Float16*)carve((size_t)DD * DD * 2);
      pO[s2][l] = (_Float16*)carve((size_t)DD * DD * 2);
    }
  _Float16* pU = (_Float16*)carve((size_t)4 * DD * DD * 2);

  auto pack = [&](const float* W, _Float16* Wp, int K) {
    int total = K * DD;
    pack_w_kernel<<<dim3((total + 255) / 256), dim3(256), 0, stream>>>(W, Wp, K, DD);
  };

  // pack all weights into WMMA B layout
  pack(peW[0], pPE[0], WDm);
  pack(peW[1], pPE[1], WDm);
  for (int s2 = 0; s2 < 2; ++s2)
    for (int l = 0; l < 2; ++l) {
      pack(Wq[s2] + (size_t)l * DD * DD, pQ[s2][l], DD);
      pack(Wk[s2] + (size_t)l * DD * DD, pK[s2][l], DD);
      pack(Wv[s2] + (size_t)l * DD * DD, pV[s2][l], DD);
      pack(Wo[s2] + (size_t)l * DD * DD, pO[s2][l], DD);
    }
  pack(updW, pU, 4 * DD);

  inv_perm_kernel<<<dim3(EN / 256), dim3(256), 0, stream>>>(perm, invp, EN);

  // zero the 16 padding rows of Q/K/V f16 buffers (e-side K padding reads them)
  zero_h_kernel<<<dim3(16), dim3(256), 0, stream>>>(Q16 + (size_t)EN * DD, 16 * DD);
  zero_h_kernel<<<dim3(16), dim3(256), 0, stream>>>(K16 + (size_t)EN * DD, 16 * DD);
  zero_h_kernel<<<dim3(16), dim3(256), 0, stream>>>(V16 + (size_t)EN * DD, 16 * DD);

  // v side (s=0): 2048 nodes x 32 ; e side (s=1): 4096 hyperedges x 16
  for (int s2 = 0; s2 < 2; ++s2) {
    const float* pefeat = (s2 == 0) ? co_in : co_con;
    const float* wfeat  = (s2 == 0) ? w_in  : w_con;
    const int nblk = (s2 == 0) ? 2048 : 4096;
    float* finalOut = (s2 == 0) ? CV : CE;

    // X = co_feat + weight @ pe_W + pe_b
    launch_gemm<F_BIAS | F_RESID, WDm>(stream, wfeat, WDm, pPE[s2],
                                       peB[s2], pefeat, X, nullptr);

    for (int l = 0; l < 2; ++l) {
      launch_gemm<F_BIAS | F_OUTF16, DD>(stream, X, DD, pQ[s2][l],
                                         Bq[s2] + l * DD, nullptr, nullptr, Q16);
      launch_gemm<F_BIAS | F_OUTF16, DD>(stream, X, DD, pK[s2][l],
                                         Bk[s2] + l * DD, nullptr, nullptr, K16);
      launch_gemm<F_BIAS | F_OUTF16, DD>(stream, X, DD, pV[s2][l],
                                         Bv[s2] + l * DD, nullptr, nullptr, V16);
      if (s2 == 0)
        attn_kernel<32><<<dim3(nblk), dim3(128), 0, stream>>>(Q16, K16, V16, Ob);
      else
        attn_kernel<16><<<dim3(nblk), dim3(128), 0, stream>>>(Q16, K16, V16, Ob);
      float* xo = (l == 1) ? finalOut : X;
      // X' = O + relu(O @ Wo + bo)
      launch_gemm<F_BIAS | F_RELU | F_RESID, DD>(stream, Ob, DD, pO[s2][l],
                                                 Bo[s2] + l * DD, Ob, xo, nullptr);
    }
  }

  // final update: out = cat(...) @ upd_W + upd_b, single fused K=1024 GEMM per output
  float* out_in  = (float*)d_out;
  float* out_con = out_in + (size_t)EN * DD;

  // out_in : [co_in(id), CV(id), CE(inv_perm), co_0(id)]
  gemm_upd_kernel<<<dim3(EN / 128, DD / 128), dim3(256), 0, stream>>>(
      co_in, nullptr, CV, nullptr, CE, invp, co_0, nullptr, pU, updb, out_in);
  // out_con: [co_in(perm), CV(perm), CE(id), co_0(perm)]
  gemm_upd_kernel<<<dim3(EN / 128, DD / 128), dim3(256), 0, stream>>>(
      co_in, perm, CV, perm, CE, nullptr, co_0, perm, pU, updb, out_con);
}
